// MultiHeadAttention_58016418234824
// MI455X (gfx1250) — matrix operand order repaired and checked by reading, not compiled, not run on hardware
//
#include <hip/hip_runtime.h>
#include <hip/hip_bf16.h>
#include <math.h>

#define SEQ  4096
#define HID  2048
#define NH   16
#define HD   128

typedef __bf16 bf16_t;
typedef __attribute__((ext_vector_type(16))) __bf16 v16bf;
typedef __attribute__((ext_vector_type(8)))  __bf16 v8bf;
typedef __attribute__((ext_vector_type(8)))  float  v8f;

static __device__ __forceinline__ v16bf cat8(v8bf lo, v8bf hi) {
  v16bf r;
#pragma unroll
  for (int i = 0; i < 8; ++i) { r[i] = lo[i]; r[i + 8] = hi[i]; }
  return r;
}

static __device__ __forceinline__ v8f wmma_bf16(v16bf a, v16bf b, v8f c) {
  // (neg_a, A, neg_b, B, c_mod, C, reuse_a, reuse_b)
  return __builtin_amdgcn_wmma_f32_16x16x32_bf16(false, a, false, b, (short)0, c,
                                                 false, false);
}

// Async global->LDS 16B copy (CDNA5 ASYNCcnt path). LDS operand is the low
// 32 bits of the generic pointer (ISA: LDS aperture keeps offset in addr[31:0]).
static __device__ __forceinline__ void async_b128(void* lds, const void* gptr) {
  unsigned l = (unsigned)(size_t)lds;
  asm volatile("global_load_async_to_lds_b128 %0, %1, off"
               :: "v"(l), "v"(gptr) : "memory");
}
static __device__ __forceinline__ void wait_async0() {
  asm volatile("s_wait_asynccnt 0x0" ::: "memory");
}

// xor-shuffle within the wave via LDS-swizzle hardware (no memory traffic).
// masks 1,2,4,8 stay inside each 16-lane half (matches C/D half-row layout).
#define SWZ_XOR(x, XOR) \
  __int_as_float(__builtin_amdgcn_ds_swizzle(__float_as_int(x), (((XOR) << 10) | 0x1F)))

// ---------------------------------------------------------------------------
// Kernel 0: fp32 -> bf16 conversion
// ---------------------------------------------------------------------------
__global__ void k_f32_to_bf16(const float* __restrict__ src,
                              bf16_t* __restrict__ dst, int n) {
  int i = blockIdx.x * blockDim.x + threadIdx.x;
  if (i < n) dst[i] = (bf16_t)src[i];
}

// ---------------------------------------------------------------------------
// Kernel 0b: per-head transpose of V:  Vt[h*HD+c][s] = V[s][h*HD+c]
// ---------------------------------------------------------------------------
__global__ void k_vtrans(const bf16_t* __restrict__ V, bf16_t* __restrict__ Vt) {
  int idx = blockIdx.x * blockDim.x + threadIdx.x;  // SEQ*HID threads
  int s  = idx & (SEQ - 1);
  int hc = idx >> 12;                               // 0..HID-1
  Vt[idx] = V[(size_t)s * HID + hc];                // coalesced store
}

// ---------------------------------------------------------------------------
// Kernel 1/4: GEMM  D[M x N] = A[M x K] * W[N x K]^T   (bf16 in, OUT out)
// block = 256 (8 waves); 128x128 tile; W tile double-buffered in LDS via
// async copies: next tile's DMA overlaps current tile's WMMAs.
// ---------------------------------------------------------------------------
template <typename OUT>
__global__ __launch_bounds__(256, 1)
void k_gemm_bt(const bf16_t* __restrict__ A,
               const bf16_t* __restrict__ W,
               OUT* __restrict__ D, int K, int N) {
  __shared__ bf16_t wt[2][128 * 32];  // W tile: [buf][n-in-tile][k-in-chunk]

  const int tid  = threadIdx.x;
  const int lane = tid & 31;
  const int wave = __builtin_amdgcn_readfirstlane(tid >> 5);
  const int half = lane >> 4;
  const int l16  = lane & 15;

  const int r0 = blockIdx.x * 128 + wave * 16;  // row tile for this wave
  const int n0 = blockIdx.y * 128;              // col tile for this block

  v8f acc[8] = {};
  const bf16_t* arow = A + (size_t)(r0 + l16) * K;

  const int wr = tid >> 1;          // 0..127: W row within tile
  const int wp = (tid & 1) * 16;    // 16-element half of the 32-wide chunk

  auto issueW = [&](int k, int buf) {
    const bf16_t* src = W + (size_t)(n0 + wr) * K + k + wp;
    bf16_t* dst = &wt[buf][wr * 32 + wp];
    async_b128(dst, src);
    async_b128(dst + 8, src + 8);
  };

  issueW(0, 0);
  wait_async0();
  __syncthreads();

  for (int k = 0; k < K; k += 32) {
    const int cur = (k >> 5) & 1;
    if (k + 32 < K) issueW(k + 32, cur ^ 1);   // prefetch next tile (DMA)

    // A fragment: row = r0+l16, K = e + 16*half (contiguous, matches B)
    v8bf alo = *(const v8bf*)(arow + k + 16 * half);
    v8bf ahi = *(const v8bf*)(arow + k + 16 * half + 8);
    v16bf afrag = cat8(alo, ahi);
#pragma unroll
    for (int nf = 0; nf < 8; ++nf) {
      // B fragment: column n = 16*nf+l16 (row of wt); K = e + 16*half
      v16bf bfrag = *(const v16bf*)&wt[cur][(16 * nf + l16) * 32 + 16 * half];
      acc[nf] = wmma_bf16(afrag, bfrag, acc[nf]);
    }

    wait_async0();     // next buffer landed (ran concurrently with WMMAs)
    __syncthreads();   // everyone done reading wt[cur]
  }

#pragma unroll
  for (int nf = 0; nf < 8; ++nf) {
#pragma unroll
    for (int j = 0; j < 8; ++j) {
      int row = r0 + j + 8 * half;     // C/D: VGPR j -> row j (+8 for hi half)
      int col = n0 + 16 * nf + l16;    // col = lane % 16
      D[(size_t)row * N + col] = (OUT)acc[nf][j];
    }
  }
}

// ---------------------------------------------------------------------------
// Kernel 2: RoPE applied in place to Q and K (bf16), llama pairing (j, j+64)
// ---------------------------------------------------------------------------
__global__ void k_rope(bf16_t* __restrict__ Q, bf16_t* __restrict__ Kb) {
  int idx = blockIdx.x * blockDim.x + threadIdx.x;   // SEQ*NH*64 threads
  int j    = idx & 63;
  int head = (idx >> 6) & (NH - 1);
  int pos  = idx >> 10;
  if (pos >= SEQ) return;

  float invf  = __powf(10000.0f, -(float)j / 64.0f);
  float ang   = (float)pos * invf;
  float s = sinf(ang), c = cosf(ang);

  size_t base = (size_t)pos * HID + head * HD;
  float q0 = (float)Q[base + j],      q1 = (float)Q[base + 64 + j];
  float k0 = (float)Kb[base + j],     k1 = (float)Kb[base + 64 + j];
  Q[base + j]       = (bf16_t)(q0 * c - q1 * s);
  Q[base + 64 + j]  = (bf16_t)(q1 * c + q0 * s);
  Kb[base + j]      = (bf16_t)(k0 * c - k1 * s);
  Kb[base + 64 + j] = (bf16_t)(k1 * c + k0 * s);
}

// ---------------------------------------------------------------------------
// Kernel 3: causal flash attention. grid = (SEQ/128, NH), block = 256.
// wave w owns query rows q0+16w..+15; 32-key steps; K tile and pre-transposed
// V tile double-buffered via async global->LDS copies (DMA overlaps WMMA).
// Scalar skip guard keeps EXEC all-1s across every WMMA.
// ---------------------------------------------------------------------------
__global__ __launch_bounds__(256, 1)
void k_attn(const bf16_t* __restrict__ Qb,
            const bf16_t* __restrict__ Kb,
            const bf16_t* __restrict__ Vt,
            bf16_t* __restrict__ Cb) {
  __shared__ bf16_t kt[2][32 * 128];   // K tile, row-major [key][d]
  __shared__ bf16_t vt[2][128 * 32];   // V tile transposed [d][key]
  __shared__ bf16_t pt[8][16 * 32];    // per-wave P tile [row][key]

  const int tid  = threadIdx.x;
  const int lane = tid & 31;
  const int wave = __builtin_amdgcn_readfirstlane(tid >> 5);  // SGPR-uniform
  const int half = lane >> 4;
  const int l16  = lane & 15;

  const int h  = blockIdx.y;
  const int q0 = blockIdx.x * 128;
  const float SCALE = 0.08838834764831845f;  // 1/sqrt(128)

  // Load Q A-fragments for this wave's 16 rows (4 chunks of K-dim 32)
  v16bf qa[4];
  {
    const bf16_t* qp = Qb + (size_t)(q0 + 16 * wave + l16) * HID + h * HD;
#pragma unroll
    for (int c = 0; c < 4; ++c) {
      v8bf lo = *(const v8bf*)(qp + c * 32 + 16 * half);
      v8bf hi = *(const v8bf*)(qp + c * 32 + 16 * half + 8);
      qa[c] = cat8(lo, hi);
    }
  }

  v8f o[8] = {};
  float rm[8], rl[8];
#pragma unroll
  for (int j = 0; j < 8; ++j) { rm[j] = -1e30f; rl[j] = 0.0f; }

  const int rhi  = q0 + 16 * wave + 15;  // scalar: last query row of this wave
  const int kend = q0 + 128;

  const int kr = tid >> 3;          // 0..31  : K-tile row
  const int kc = (tid & 7) * 16;    // 0..112 : 16-elem column chunk
  const int vr = tid >> 1;          // 0..127 : Vt row (d index)
  const int vp = (tid & 1) * 16;    // 0/16   : key chunk

  auto issueKV = [&](int kk, int buf) {
    const bf16_t* ks = Kb + (size_t)(kk + kr) * HID + h * HD + kc;
    bf16_t* kd = &kt[buf][kr * 128 + kc];
    async_b128(kd, ks);
    async_b128(kd + 8, ks + 8);
    const bf16_t* vs = Vt + (size_t)(h * HD + vr) * SEQ + kk + vp;
    bf16_t* vd = &vt[buf][vr * 32 + vp];
    async_b128(vd, vs);
    async_b128(vd + 8, vs + 8);
  };

  issueKV(0, 0);
  wait_async0();
  __syncthreads();

  for (int kk = 0; kk < kend; kk += 32) {
    const int cur = (kk >> 5) & 1;
    if (kk + 32 < kend) issueKV(kk + 32, cur ^ 1);  // prefetch next tiles

    if (kk <= rhi) {  // scalar condition -> s_cbranch, EXEC untouched
      // ---- S = Q * K^T over two 16-key subtiles ----
      v8f s0 = {}, s1 = {};
#pragma unroll
      for (int c = 0; c < 4; ++c) {
        v16bf b0 = *(const v16bf*)&kt[cur][l16 * 128 + c * 32 + 16 * half];
        v16bf b1 = *(const v16bf*)&kt[cur][(16 + l16) * 128 + c * 32 + 16 * half];
        s0 = wmma_bf16(qa[c], b0, s0);
        s1 = wmma_bf16(qa[c], b1, s1);
      }

      // ---- causal mask + online softmax ----
      float f0[8], f1[8];
      const int rbase = q0 + 16 * wave + 8 * half;
      const int c0 = kk + l16, c1 = kk + 16 + l16;
#pragma unroll
      for (int j = 0; j < 8; ++j) {
        int row = rbase + j;
        f0[j] = (c0 <= row) ? s0[j] * SCALE : -1e30f;
        f1[j] = (c1 <= row) ? s1[j] * SCALE : -1e30f;
      }
#pragma unroll
      for (int j = 0; j < 8; ++j) {
        float x = fmaxf(f0[j], f1[j]);
        x = fmaxf(x, SWZ_XOR(x, 1));
        x = fmaxf(x, SWZ_XOR(x, 2));
        x = fmaxf(x, SWZ_XOR(x, 4));
        x = fmaxf(x, SWZ_XOR(x, 8));
        float mnew  = fmaxf(rm[j], x);
        float alpha = __expf(rm[j] - mnew);
        rm[j] = mnew;
        float p0 = __expf(f0[j] - mnew);
        float p1 = __expf(f1[j] - mnew);
        float rs = p0 + p1;
        rs += SWZ_XOR(rs, 1);
        rs += SWZ_XOR(rs, 2);
        rs += SWZ_XOR(rs, 4);
        rs += SWZ_XOR(rs, 8);
        rl[j] = rl[j] * alpha + rs;
#pragma unroll
        for (int nf = 0; nf < 8; ++nf) o[nf][j] *= alpha;
        pt[wave][(j + 8 * half) * 32 + l16]      = (bf16_t)p0;
        pt[wave][(j + 8 * half) * 32 + 16 + l16] = (bf16_t)p1;
      }

      // ---- O += P * V (P re-laid out via LDS into A fragment) ----
      const bf16_t* pr = &pt[wave][l16 * 32];
      v8bf plo = *(const v8bf*)(pr + 16 * half);
      v8bf phi = *(const v8bf*)(pr + 16 * half + 8);
      v16bf pfrag = cat8(plo, phi);
#pragma unroll
      for (int nf = 0; nf < 8; ++nf) {
        v16bf vb = *(const v16bf*)&vt[cur][(nf * 16 + l16) * 32 + 16 * half];
        o[nf] = wmma_bf16(pfrag, vb, o[nf]);
      }
    }

    wait_async0();     // prefetched tiles landed (overlapped with compute)
    __syncthreads();   // everyone done reading buffers[cur]
  }

  // ---- normalize and write context (bf16) ----
#pragma unroll
  for (int j = 0; j < 8; ++j) {
    float inv = 1.0f / rl[j];
    int row = q0 + 16 * wave + 8 * half + j;
#pragma unroll
    for (int nf = 0; nf < 8; ++nf) {
      Cb[(size_t)row * HID + h * HD + nf * 16 + l16] = (bf16_t)(o[nf][j] * inv);
    }
  }
}

// ---------------------------------------------------------------------------
extern "C" void kernel_launch(void* const* d_in, const int* in_sizes, int n_in,
                              void* d_out, int out_size, void* d_ws, size_t ws_size,
                              hipStream_t stream) {
  const float* X  = (const float*)d_in[0];
  const float* Wq = (const float*)d_in[1];
  const float* Wk = (const float*)d_in[2];
  const float* Wv = (const float*)d_in[3];
  const float* Wo = (const float*)d_in[4];
  float* out = (float*)d_out;

  char* ws = (char*)d_ws;
  size_t off = 0;
  const size_t SH = (size_t)SEQ * HID * sizeof(bf16_t);
  const size_t WW = (size_t)HID * HID * sizeof(bf16_t);
  bf16_t* Xb  = (bf16_t*)(ws + off); off += SH;
  bf16_t* Wqb = (bf16_t*)(ws + off); off += WW;
  bf16_t* Wkb = (bf16_t*)(ws + off); off += WW;
  bf16_t* Wvb = (bf16_t*)(ws + off); off += WW;
  bf16_t* Wob = (bf16_t*)(ws + off); off += WW;
  bf16_t* Qb  = (bf16_t*)(ws + off); off += SH;
  bf16_t* Kb  = (bf16_t*)(ws + off); off += SH;
  bf16_t* Vb  = (bf16_t*)(ws + off); off += SH;
  bf16_t* Vtb = (bf16_t*)(ws + off); off += SH;
  bf16_t* Cb  = (bf16_t*)(ws + off); off += SH;

  const int nX = SEQ * HID, nW = HID * HID;
  k_f32_to_bf16<<<(nX + 255) / 256, 256, 0, stream>>>(X,  Xb,  nX);
  k_f32_to_bf16<<<(nW + 255) / 256, 256, 0, stream>>>(Wq, Wqb, nW);
  k_f32_to_bf16<<<(nW + 255) / 256, 256, 0, stream>>>(Wk, Wkb, nW);
  k_f32_to_bf16<<<(nW + 255) / 256, 256, 0, stream>>>(Wv, Wvb, nW);
  k_f32_to_bf16<<<(nW + 255) / 256, 256, 0, stream>>>(Wo, Wob, nW);

  dim3 gGemm(SEQ / 128, HID / 128);
  k_gemm_bt<bf16_t><<<gGemm, 256, 0, stream>>>(Xb, Wqb, Qb, HID, HID);
  k_gemm_bt<bf16_t><<<gGemm, 256, 0, stream>>>(Xb, Wkb, Kb, HID, HID);
  k_gemm_bt<bf16_t><<<gGemm, 256, 0, stream>>>(Xb, Wvb, Vb, HID, HID);

  int nR = SEQ * NH * 64;
  k_rope<<<(nR + 255) / 256, 256, 0, stream>>>(Qb, Kb);

  k_vtrans<<<(nX + 255) / 256, 256, 0, stream>>>(Vb, Vtb);

  k_attn<<<dim3(SEQ / 128, NH), 256, 0, stream>>>(Qb, Kb, Vtb, Cb);

  k_gemm_bt<float><<<gGemm, 256, 0, stream>>>(Cb, Wob, out, HID, HID);
}
